// gru_model_89283780149395
// MI455X (gfx1250) — matrix operand order repaired and checked by reading, not compiled, not run on hardware
//
#include <hip/hip_runtime.h>

// ---- problem constants (from reference) ----
#define HDIM  1024
#define SEQ   512
#define BATCH 64
#define G3    3072   // 3*H
#define NWG   32     // workgroups in the persistent scan kernel (32 cols each)

typedef __attribute__((ext_vector_type(16))) __bf16 v16bf;
typedef __attribute__((ext_vector_type(8)))  __bf16 v8bf;
typedef __attribute__((ext_vector_type(8)))  float  v8f;

// ---------- helpers ----------
__device__ __forceinline__ float sigmoid_(float x) {
    return 1.0f / (1.0f + __expf(-x));
}
__device__ __forceinline__ float tanh_(float x) {
    x = fminf(fmaxf(x, -20.0f), 20.0f);
    float e = __expf(2.0f * x);
    return (e - 1.0f) / (e + 1.0f);
}

// Device-scope barrier across the NWG resident workgroups of the scan kernel.
// Monotonic counter: after step t every block adds 1; proceed when >= NWG*(t+1).
__device__ __forceinline__ void grid_barrier_(volatile unsigned int* bar,
                                              unsigned int target) {
    __threadfence();              // each thread releases its global stores
    __syncthreads();
    if (threadIdx.x == 0) {
        atomicAdd((unsigned int*)bar, 1u);
        while (*bar < target) { __builtin_amdgcn_s_sleep(1); }
        __threadfence();          // acquire: discard stale cached lines
    }
    __syncthreads();
}

// ---------- kernel 0: transpose + convert weights to bf16, N-major ----------
// W: [H, 3H] row-major fp32  ->  WT: [3H, H] row-major bf16
__global__ __launch_bounds__(256) void wt_bf16_kernel(const float* __restrict__ W,
                                                      __bf16* __restrict__ WT) {
    size_t i = (size_t)blockIdx.x * blockDim.x + threadIdx.x;
    size_t total = (size_t)HDIM * G3;
    if (i >= total) return;
    int n = (int)(i / HDIM);   // output row    (0..3H)
    int k = (int)(i % HDIM);   // output column (0..H)   -> coalesced store
    WT[(size_t)n * HDIM + k] = (__bf16)W[(size_t)k * G3 + n];
}

// ---------- kernel: zero-init h0 + barrier counter (ws is poisoned) ----------
__global__ __launch_bounds__(256) void h0_init_kernel(float* __restrict__ hf,
                                                      __bf16* __restrict__ hb,
                                                      unsigned int* __restrict__ bar) {
    int i = blockIdx.x * blockDim.x + threadIdx.x;
    if (i < BATCH * HDIM) { hf[i] = 0.0f; hb[i] = (__bf16)0.0f; }
    if (i == 0) *bar = 0u;
}

// ---------- kernel 1: gi_all = gather(emb, ids) @ W_ih + b_ih  (bf16 WMMA) ----------
// M = SEQ*BATCH = 32768, N = 3072, K = 1024. Each wave computes a 16x64 strip.
__global__ __launch_bounds__(256) void gi_gemm_kernel(const int* __restrict__ ids,
                                                      const float* __restrict__ emb,
                                                      const __bf16* __restrict__ WihT,
                                                      const float* __restrict__ b_ih,
                                                      __bf16* __restrict__ gi) {
    const int lane  = threadIdx.x & 31;
    const int wid   = blockIdx.x * 8 + (threadIdx.x >> 5);
    const int m_tile = wid / 48;          // 0..2047
    const int n64    = wid % 48;          // 0..47 (64-wide N strip)
    const int half  = lane >> 4;
    const int r     = lane & 15;

    const int row = m_tile * 16 + r;      // (s*BATCH + b) flat row
    const int tok = ids[row];
    const float* arow = emb + (size_t)tok * HDIM;

    v8f acc[4] = {};
    for (int k0 = 0; k0 < HDIM; k0 += 32) {
        v16bf a;
        const float* p0 = arow + k0 + half * 16;
#pragma unroll
        for (int i = 0; i < 8; ++i) {     // fp32 -> bf16 on the fly
            a[i]     = (__bf16)p0[i];
            a[i + 8] = (__bf16)p0[i + 8];
        }
#pragma unroll
        for (int j = 0; j < 4; ++j) {
            const int n = n64 * 64 + j * 16 + r;                 // B lane column
            const v16bf b = *(const v16bf*)(WihT + (size_t)n * HDIM + k0 + half * 16);
            acc[j] = __builtin_amdgcn_wmma_f32_16x16x32_bf16(
                false, a, false, b, (short)0, acc[j], false, false);
        }
    }
#pragma unroll
    for (int j = 0; j < 4; ++j) {
        const int col  = n64 * 64 + j * 16 + r;
        const float bias = b_ih[col];
#pragma unroll
        for (int v = 0; v < 8; ++v) {     // C/D layout: M = half*8 + v, N = r
            const int orow = m_tile * 16 + half * 8 + v;
            gi[(size_t)orow * G3 + col] = (__bf16)(acc[j][v] + bias);
        }
    }
}

// ---------- kernel 2: persistent GRU scan (all 512 steps, one launch) ----------
// NWG=32 workgroups; workgroup wg owns hidden columns [wg*32, wg*32+32).
// Its W_hh slice (3 gates x 32 cols x 1024 K, bf16 = 192 KB) is loaded into the
// 320 KB LDS ONCE; the per-step WMMA hot loop feeds B from ds_load instead of L2.
__global__ __launch_bounds__(256) void gru_scan_kernel(const __bf16* __restrict__ gi,
                                                       const __bf16* __restrict__ WhhT,
                                                       const float* __restrict__ b_hh,
                                                       float*  __restrict__ hF0,
                                                       float*  __restrict__ hF1,
                                                       __bf16* __restrict__ hB0,
                                                       __bf16* __restrict__ hB1,
                                                       unsigned int* __restrict__ bar) {
    __shared__ __bf16 Wlds[96 * HDIM];    // 192 KB: [c][k], c = gate*32 + j
    __shared__ float  gh[BATCH * 96];     //  24 KB: [m][c]

    const int wg    = blockIdx.x;         // 0..NWG-1
    const int lane  = threadIdx.x & 31;
    const int wave  = threadIdx.x >> 5;   // 0..7
    const int m_tile = wave & 3;          // 4 M-tiles cover batch 64
    const int ngrp   = wave >> 2;         // 2 groups x 3 N-tiles cover 96 cols
    const int half  = lane >> 4;
    const int r     = lane & 15;

    // ---- one-time LDS fill of this workgroup's W_hh slice (coalesced 16B) ----
    for (int f = threadIdx.x; f < 96 * (HDIM / 8); f += 256) {
        const int c  = f >> 7;            // 0..95
        const int kc = (f & 127) * 8;     // 0..1016 step 8
        const int gate = c >> 5;
        const int jj   = c & 31;
        *(v8bf*)&Wlds[c * HDIM + kc] =
            *(const v8bf*)(WhhT + (size_t)(gate * HDIM + wg * 32 + jj) * HDIM + kc);
    }
    __syncthreads();

    for (int t = 0; t < SEQ; ++t) {
        const __bf16* h_prev_b = (t & 1) ? hB1 : hB0;
        const float*  h_prev_f = (t & 1) ? hF1 : hF0;
        __bf16*       h_next_b = (t & 1) ? hB0 : hB1;
        float*        h_next_f = (t & 1) ? hF0 : hF1;
        const __bf16* gi_step  = gi + (size_t)t * BATCH * G3;

        const __bf16* arow = h_prev_b + (size_t)(m_tile * 16 + r) * HDIM;

        v8f acc[3] = {};
        for (int k0 = 0; k0 < HDIM; k0 += 32) {
            v16bf a;
            const __bf16* p0 = arow + k0 + half * 16;
#pragma unroll
            for (int i = 0; i < 8; ++i) { a[i] = p0[i]; a[i + 8] = p0[i + 8]; }
#pragma unroll
            for (int tt = 0; tt < 3; ++tt) {
                const int c = (ngrp * 3 + tt) * 16 + r;   // local column 0..95
                const v16bf b = *(const v16bf*)&Wlds[c * HDIM + k0 + half * 16];
                acc[tt] = __builtin_amdgcn_wmma_f32_16x16x32_bf16(
                    false, a, false, b, (short)0, acc[tt], false, false);
            }
        }
#pragma unroll
        for (int tt = 0; tt < 3; ++tt) {
            const int c = (ngrp * 3 + tt) * 16 + r;
#pragma unroll
            for (int v = 0; v < 8; ++v) {
                const int m = m_tile * 16 + half * 8 + v;
                gh[m * 96 + c] = acc[tt][v];
            }
        }
        __syncthreads();

        for (int e = threadIdx.x; e < BATCH * 32; e += 256) {
            const int m   = e >> 5;
            const int j   = e & 31;
            const int col = wg * 32 + j;
            const float ghr = gh[m * 96 + j]       + b_hh[col];
            const float ghz = gh[m * 96 + 32 + j]  + b_hh[HDIM + col];
            const float ghn = gh[m * 96 + 64 + j]  + b_hh[2 * HDIM + col];
            const float gir = (float)gi_step[(size_t)m * G3 + col];
            const float giz = (float)gi_step[(size_t)m * G3 + HDIM + col];
            const float gin = (float)gi_step[(size_t)m * G3 + 2 * HDIM + col];
            const float rg = sigmoid_(gir + ghr);
            const float zg = sigmoid_(giz + ghz);
            const float ng = tanh_(gin + rg * ghn);
            const float hp = h_prev_f[(size_t)m * HDIM + col];
            const float hn = (1.0f - zg) * ng + zg * hp;
            h_next_f[(size_t)m * HDIM + col] = hn;
            h_next_b[(size_t)m * HDIM + col] = (__bf16)hn;
        }

        // warm caches with next step's gi slice while other workgroups finish
        if (t + 1 < SEQ) {
            const __bf16* gnext = gi + (size_t)(t + 1) * BATCH * G3;
            for (int e = threadIdx.x; e < BATCH * 32; e += 256) {
                const int m = e >> 5;
                const int j = e & 31;
                __builtin_prefetch(gnext + (size_t)m * G3 + wg * 32 + j, 0, 1);
            }
        }

        grid_barrier_(bar, (unsigned int)(NWG * (t + 1)));
    }
}

// ---------- kernel 3: decode head (tiny, one workgroup) ----------
__global__ __launch_bounds__(256) void decode_kernel(const float* __restrict__ hf,
                                                     const float* __restrict__ W_dec,
                                                     const float* __restrict__ b_dec,
                                                     const float* __restrict__ W_fc,
                                                     const float* __restrict__ b_fc,
                                                     float* __restrict__ out) {
    __shared__ float dec[BATCH * 100];
    for (int d = threadIdx.x; d < BATCH * 100; d += 256) {
        const int m = d / 100, c = d % 100;
        float s = b_dec[c];
        for (int k = 0; k < HDIM; ++k) s += hf[m * HDIM + k] * W_dec[k * 100 + c];
        dec[d] = fmaxf(s, 0.0f);
    }
    __syncthreads();
    if (threadIdx.x < BATCH * 2) {
        const int m = threadIdx.x >> 1, o = threadIdx.x & 1;
        float s = b_fc[o];
        for (int c = 0; c < 100; ++c) s += dec[m * 100 + c] * W_fc[c * 2 + o];
        out[threadIdx.x] = s;
    }
}

// ---------- host ----------
extern "C" void kernel_launch(void* const* d_in, const int* in_sizes, int n_in,
                              void* d_out, int out_size, void* d_ws, size_t ws_size,
                              hipStream_t stream) {
    const int*   ids   = (const int*)  d_in[0];
    const float* emb   = (const float*)d_in[1];
    const float* W_ih  = (const float*)d_in[2];
    const float* W_hh  = (const float*)d_in[3];
    const float* b_ih  = (const float*)d_in[4];
    const float* b_hh  = (const float*)d_in[5];
    const float* W_dec = (const float*)d_in[6];
    const float* b_dec = (const float*)d_in[7];
    const float* W_fc  = (const float*)d_in[8];
    const float* b_fc  = (const float*)d_in[9];

    // workspace carve-up (~215 MB total)
    char* ws = (char*)d_ws;
    __bf16* WihT = (__bf16*)ws;  ws += (size_t)G3 * HDIM * 2;     // 6 MB
    __bf16* WhhT = (__bf16*)ws;  ws += (size_t)G3 * HDIM * 2;     // 6 MB
    float*  hF[2];
    hF[0] = (float*)ws;  ws += (size_t)BATCH * HDIM * 4;
    hF[1] = (float*)ws;  ws += (size_t)BATCH * HDIM * 4;
    __bf16* hB[2];
    hB[0] = (__bf16*)ws; ws += (size_t)BATCH * HDIM * 2;
    hB[1] = (__bf16*)ws; ws += (size_t)BATCH * HDIM * 2;
    unsigned int* bar = (unsigned int*)ws;  ws += 256;            // barrier counter
    __bf16* gi = (__bf16*)ws;   // SEQ*BATCH*3H bf16 = 201 MB

    // 1) weights -> bf16, N-major (WMMA B layout wants contiguous K per column)
    {
        size_t total = (size_t)HDIM * G3;
        int blocks = (int)((total + 255) / 256);
        wt_bf16_kernel<<<blocks, 256, 0, stream>>>(W_ih, WihT);
        wt_bf16_kernel<<<blocks, 256, 0, stream>>>(W_hh, WhhT);
    }

    // 2) h0 = 0 (both precisions) + reset barrier counter
    h0_init_kernel<<<(BATCH * HDIM + 255) / 256, 256, 0, stream>>>(hF[0], hB[0], bar);

    // 3) big parallel GEMM: gi_all (off the serial critical path)
    gi_gemm_kernel<<<(2048 * 48) / 8, 256, 0, stream>>>(ids, emb, WihT, b_ih, gi);

    // 4) all 512 GRU steps in ONE persistent launch; W_hh lives in LDS (192 KB
    //    of the 320 KB WGP LDS), device-scope atomic barrier between steps
    gru_scan_kernel<<<NWG, 256, 0, stream>>>(gi, WhhT, b_hh,
                                             hF[0], hF[1], hB[0], hB[1], bar);

    // 5) decode head; final h is in buffer 0 (512 is even)
    decode_kernel<<<1, 256, 0, stream>>>(hF[0], W_dec, b_dec, W_fc, b_fc, (float*)d_out);
}
